// _ConvBNReLU_53584011985474
// MI455X (gfx1250) — matrix operand order repaired and checked by reading, not compiled, not run on hardware
//
#include <hip/hip_runtime.h>

typedef _Float16 h8  __attribute__((ext_vector_type(8)));
typedef _Float16 h16 __attribute__((ext_vector_type(16)));
typedef float    f8  __attribute__((ext_vector_type(8)));

#define CIN   64
#define COUT  64
#define HH    256
#define WW    256
#define KTOT  576        // CIN * 9, GEMM K
#define ROWS  4          // output rows per workgroup
#define XR    (ROWS + 2) // staged input rows (with halo)
#define XC    66         // staged input cols (64 + halo)
#define CPAD  72         // padded cin stride in LDS (halves): 144B, 16B-aligned, low bank conflict

#if __has_builtin(__builtin_amdgcn_sched_barrier)
#define SCHED_FENCE() __builtin_amdgcn_sched_barrier(0)
#else
#define SCHED_FENCE()
#endif

// ---------------- weight fake-quant prep: one block per output channel ----------------
__global__ __launch_bounds__(128)
void wprep_kernel(const float* __restrict__ Wsrc, _Float16* __restrict__ W16) {
  __shared__ float red[128];
  const int co = blockIdx.x;
  const float* wrow = Wsrc + (size_t)co * KTOT;
  float m = 0.0f;
  for (int i = threadIdx.x; i < KTOT; i += 128)
    m = fmaxf(m, fabsf(wrow[i]));
  red[threadIdx.x] = m;
  __syncthreads();
  for (int s = 64; s > 0; s >>= 1) {
    if (threadIdx.x < s) red[threadIdx.x] = fmaxf(red[threadIdx.x], red[threadIdx.x + s]);
    __syncthreads();
  }
  const float scale = fmaxf(red[0] / 7.0f, 1e-8f);
  for (int i = threadIdx.x; i < KTOT; i += 128) {
    float q = rintf(wrow[i] / scale);              // RNE, matches jnp.round
    q = fminf(fmaxf(q, -7.0f), 7.0f) * scale;
    const int cin = i / 9, tap = i - cin * 9;      // src layout [cout][cin][kh][kw]
    W16[(size_t)co * KTOT + tap * 64 + cin] = (_Float16)q;  // K = tap*64 + cin
  }
}

// ---------------- main implicit-GEMM conv + BN + QuantReLU ----------------
__global__ __launch_bounds__(256)
__attribute__((amdgpu_waves_per_eu(4)))   // cap at 256 VGPRs: no MSB-bank toggling, no spills
void convmain_kernel(const float* __restrict__ x, const _Float16* __restrict__ W16,
                     const float* __restrict__ bn_g, const float* __restrict__ bn_b,
                     const float* __restrict__ bn_m, const float* __restrict__ bn_v,
                     const float* __restrict__ act_s, float* __restrict__ out) {
  __shared__ _Float16 sx[XR * XC * CPAD];   // 6*66*72 halves = 57,024 B

  const int tid  = threadIdx.x;
  const int lane = tid & 31;
  const int wv   = tid >> 5;
  const int mi   = wv >> 1;                 // channel block 0..3
  const int ni   = wv & 1;                  // pixel half 0..1
  const int wbase = blockIdx.x * 64;
  const int h0    = blockIdx.y * ROWS;
  const int b     = blockIdx.z;

  // ---- stage x tile f32 -> f16 into LDS, layout [row][col][cin] ----
  // 384 (row,cin) pairs split across 8 waves; lanes sweep cols (coalesced global reads)
  for (int p = wv * 48; p < wv * 48 + 48; ++p) {
    const int rr  = p >> 6;
    const int cin = p & 63;
    const int hin = h0 - 1 + rr;
    const bool hok = (hin >= 0) & (hin < HH);
    const float* src = x + (((size_t)b * CIN + cin) * HH + hin) * WW + (wbase - 1);
    for (int c = lane; c < XC; c += 32) {
      const int wg = wbase - 1 + c;
      const float v = (hok && wg >= 0 && wg < WW) ? src[c] : 0.0f;
      sx[(rr * XC + c) * CPAD + cin] = (_Float16)v;
    }
  }
  __syncthreads();

  f8 acc[ROWS][2];
  const f8 fz = {};
#pragma unroll
  for (int r = 0; r < ROWS; ++r) { acc[r][0] = fz; acc[r][1] = fz; }

  // Lane layouts used here (contiguous k, same for A and B; not AMD's table order):
  // A (16x32 f16): row M = lane&15; 16 consecutive halves at K0 = s*32 + 16*(lane>=16)
  // B (32x16 f16): col N = lane&15; 16 consecutive halves at K = 16*(lane>=16)
  const int mrow    = mi * 16 + (lane & 15);
  const _Float16* wrow = W16 + (size_t)mrow * KTOT;
  const int laneHiA = (lane >> 4) << 4;   // +16 halves
  const int laneHiB = (lane >> 4) << 4;   // +16 halves
  const int ncol    = ni * 32 + (lane & 15);

#pragma unroll 1   // keep K loop rolled (bounded registers); pipeline within the body
  for (int slice = 0; slice < 18; ++slice) {
    const int tap = slice >> 1;
    const int kh  = tap / 3;
    const int kw  = tap - kh * 3;
    const int cinHalf = (slice & 1) * 32;

    SCHED_FENCE();   // keep the load burst together, after the previous WMMA burst

    // A fragment first: global latency overlaps the DS burst below
    const _Float16* ap = wrow + slice * 32 + laneHiA;
    h16 A;
    *((h8*)&A)       = *(const h8*)(ap);
    *(((h8*)&A) + 1) = *(const h8*)(ap + 8);

    // All 16 ds_load_b128 issued into DISTINCT live values...
    h16 Bf[ROWS][2];
#pragma unroll
    for (int r = 0; r < ROWS; ++r) {
#pragma unroll
      for (int nt = 0; nt < 2; ++nt) {
        const int col = ncol + nt * 16 + kw;
        const _Float16* bp = &sx[((r + kh) * XC + col) * CPAD + cinHalf + laneHiB];
        *((h8*)&Bf[r][nt])       = *(const h8*)(bp);      // ds_load_b128
        *(((h8*)&Bf[r][nt]) + 1) = *(const h8*)(bp + 8);  // ds_load_b128
      }
    }

    SCHED_FENCE();   // ...scheduler may NOT sink loads past this point:
                     // forces 8 live B buffers; DS returns in order, so each WMMA
                     // gets a partial s_wait_dscnt (14,12,10,...) instead of 0.

#pragma unroll
    for (int r = 0; r < ROWS; ++r) {
#pragma unroll
      for (int nt = 0; nt < 2; ++nt) {
        acc[r][nt] = __builtin_amdgcn_wmma_f32_16x16x32_f16(
            false, A, false, Bf[r][nt], (short)0, acc[r][nt], false, false);
      }
    }
  }

  // ---- epilogue: BN fold + ReLU + uint4 fake-quant ----
  const float s     = fmaxf(act_s[0], 1e-8f);
  const float inv_s = 1.0f / s;
  float bna[8], bnb[8];
  const int cbase = mi * 16 + ((lane >> 4) << 3);
#pragma unroll
  for (int vr = 0; vr < 8; ++vr) {
    const int c = cbase + vr;                       // C/D layout: M = vr + 8*(lane>=16)
    const float a = bn_g[c] * (1.0f / sqrtf(bn_v[c] + 1e-5f));
    bna[vr] = a;
    bnb[vr] = bn_b[c] - bn_m[c] * a;
  }

#pragma unroll
  for (int r = 0; r < ROWS; ++r) {
    const int h = h0 + r;
#pragma unroll
    for (int nt = 0; nt < 2; ++nt) {
      const int wcol = wbase + ni * 32 + nt * 16 + (lane & 15);
#pragma unroll
      for (int vr = 0; vr < 8; ++vr) {
        const int c = cbase + vr;
        float y = acc[r][nt][vr] * bna[vr] + bnb[vr];
        y = fmaxf(y, 0.0f);
        y = fminf(fmaxf(rintf(y * inv_s), 0.0f), 15.0f) * s;
        __builtin_nontemporal_store(
            y, &out[(((size_t)b * COUT + c) * HH + h) * WW + wcol]);
      }
    }
  }
}

extern "C" void kernel_launch(void* const* d_in, const int* in_sizes, int n_in,
                              void* d_out, int out_size, void* d_ws, size_t ws_size,
                              hipStream_t stream) {
  const float* x    = (const float*)d_in[0];
  const float* Wt   = (const float*)d_in[1];
  const float* bn_g = (const float*)d_in[2];
  const float* bn_b = (const float*)d_in[3];
  const float* bn_m = (const float*)d_in[4];
  const float* bn_v = (const float*)d_in[5];
  const float* as_  = (const float*)d_in[6];
  float* out = (float*)d_out;

  _Float16* W16 = (_Float16*)d_ws;   // 64*576*2 = 73,728 B of scratch

  wprep_kernel<<<COUT, 128, 0, stream>>>(Wt, W16);

  dim3 grid(WW / 64, HH / ROWS, 16);
  convmain_kernel<<<grid, 256, 0, stream>>>(x, W16, bn_g, bn_b, bn_m, bn_v, as_, out);
}
